// DecoupleModelOnlyWang_etal_5420248727873
// MI455X (gfx1250) — matrix operand order repaired and checked by reading, not compiled, not run on hardware
//
#include <hip/hip_runtime.h>
#include <hip/hip_bf16.h>

typedef __attribute__((ext_vector_type(16))) _Float16 v16h;
typedef __attribute__((ext_vector_type(8)))  float    v8f;
typedef int v4i __attribute__((vector_size(16)));

#define BSZ 16
#define HH 320
#define WW 320
#define HW (HH*WW)
#define EPSV 1e-5f

#ifndef __has_builtin
#define __has_builtin(x) 0
#endif

#define GLOBAL_AS __attribute__((address_space(1)))
#define LOCAL_AS  __attribute__((address_space(3)))

#if __has_builtin(__builtin_amdgcn_global_load_async_to_lds_b128)
#define HAVE_ASYNC_LDS 1
// (global src v4i*, lds dst v4i*, imm offset, imm cpol)
#define ASYNC_CP128(gsrc, ldst)                                            \
    __builtin_amdgcn_global_load_async_to_lds_b128(                        \
        (GLOBAL_AS v4i*)(v4i*)(gsrc), (LOCAL_AS v4i*)(v4i*)(ldst), 0, 0)
#else
#define HAVE_ASYNC_LDS 0
#endif

#if __has_builtin(__builtin_amdgcn_s_wait_asynccnt)
#define ASYNC_WAIT() __builtin_amdgcn_s_wait_asynccnt(0)
#else
#define ASYNC_WAIT() asm volatile("s_wait_asynccnt 0x0" ::: "memory")
#endif

// ---------------------------------------------------------------------------
// Weight generation: w = acc[b]*W + b  (tiny affine hypernetwork)
// ---------------------------------------------------------------------------
__global__ void gen_small_kernel(const float* __restrict__ acc,
                                 const float* __restrict__ W1, const float* __restrict__ b1,
                                 const float* __restrict__ W5, const float* __restrict__ b5,
                                 float* __restrict__ w1f, float* __restrict__ w5f) {
    int i = blockIdx.x * 256 + threadIdx.x;
    if (i < BSZ * 288) {
        int b = i / 288, idx = i % 288;            // idx = oc*9 + t   (32,1,3,3)
        w1f[i] = acc[b] * W1[idx] + b1[idx];
    } else if (i < 2 * BSZ * 288) {
        int j = i - BSZ * 288;
        int b = j / 288, idx = j % 288;            // idx = ic*9 + t   (1,32,3,3)
        w5f[j] = acc[b] * W5[idx] + b5[idx];
    }
}

// Mid-layer weights pre-swizzled into WMMA B-fragment layout (f16 16x16x32):
// lane L<16: N=L, K={0..7,16..23} ; lane L>=16: N=L-16, K={8..15,24..31}
__global__ void gen_frag_kernel(const float* __restrict__ acc,
                                const float* __restrict__ Wmid, const float* __restrict__ bmid,
                                _Float16* __restrict__ wfrag) {
    int i = blockIdx.x * 256 + threadIdx.x;
    if (i >= 3 * BSZ * 9216) return;
    int rest = i % 9216;
    int lb = i / 9216;
    int b = lb % BSZ;
    int l = lb / BSZ;
    int kk = rest & 15;
    int L  = (rest >> 4) & 31;
    int th = rest >> 9;                 // t*2+h  (0..17)
    int t = th >> 1, h = th & 1;
    int n = (L & 15) + 16 * h;
    int c = (kk & 7) + ((kk >> 3) << 4) + ((L >> 4) << 3);
    int widx = (n * 32 + c) * 9 + t;    // (32,32,3,3) flat
    float v = acc[b] * Wmid[l * 9216 + widx] + bmid[l * 9216 + widx];
    wfrag[i] = (_Float16)v;
}

// ---------------------------------------------------------------------------
// Layer 1: 1->32 direct conv (K=9, too small for WMMA).
// ---------------------------------------------------------------------------
__global__ __launch_bounds__(256) void conv1_kernel(const float* __restrict__ x,
                                                    _Float16* __restrict__ actOut,
                                                    const float* __restrict__ w1f,
                                                    float* __restrict__ stats) {
    __shared__ float sW[288];
    __shared__ float sStats[64];
    int b = blockIdx.y;
    int tid = threadIdx.x;
    for (int i = tid; i < 288; i += 256) sW[i] = w1f[b * 288 + i];
    if (tid < 64) sStats[tid] = 0.f;
    __syncthreads();

    int p = blockIdx.x * 256 + tid;
    int yy = p / WW, xx = p % WW;
    float xin[9];
#pragma unroll
    for (int t = 0; t < 9; ++t) {
        int dy = t / 3, dx = t % 3;
        int y2 = yy + dy - 1, x2 = xx + dx - 1;
        xin[t] = (y2 >= 0 && y2 < HH && x2 >= 0 && x2 < WW)
                 ? x[(size_t)b * HW + y2 * WW + x2] : 0.f;
    }
    union { _Float16 h[32]; uint4 q[4]; } u;
#pragma unroll
    for (int oc = 0; oc < 32; ++oc) {
        float a = 0.f;
#pragma unroll
        for (int t = 0; t < 9; ++t) a += xin[t] * sW[oc * 9 + t];
        u.h[oc] = (_Float16)a;
        atomicAdd(&sStats[oc], a);
        atomicAdd(&sStats[32 + oc], a * a);
    }
    uint4* dst = (uint4*)(actOut + ((size_t)b * HW + p) * 32);
    dst[0] = u.q[0]; dst[1] = u.q[1]; dst[2] = u.q[2]; dst[3] = u.q[3];
    __syncthreads();
    if (tid < 64) atomicAdd(&stats[b * 64 + tid], sStats[tid]);
}

// ---------------------------------------------------------------------------
// Mid layers: 32->32 implicit GEMM with v_wmma_f32_16x16x32_f16.
// Staging via GLOBAL_LOAD_ASYNC_TO_LDS_B128 when the toolchain exposes it.
// ---------------------------------------------------------------------------
#define TILE_R 8
#define TILE_C 16
#define HALO_R 10
#define HALO_C 18

__global__ __launch_bounds__(256) void convmid_kernel(const _Float16* __restrict__ actIn,
                                                      _Float16* __restrict__ actOut,
                                                      const _Float16* __restrict__ wfragL,
                                                      float* __restrict__ stats) {
    __shared__ _Float16 sW[9216];                     // 18 KB weight fragments
    __shared__ _Float16 sA[HALO_R * HALO_C * 32];     // 11.5 KB halo, channel-last
    __shared__ float    sOut[TILE_R * TILE_C * 32];   // 16 KB f32 staging
    __shared__ float    sStats[64];

    int b  = blockIdx.z;
    int ty = blockIdx.y;     // 0..39
    int tx = blockIdx.x;     // 0..19
    int tid = threadIdx.x;

    // stage weights for this sample (contiguous 18KB)
    uint4* wsrc = (uint4*)(wfragL + (size_t)b * 9216);
    uint4* sW4w = (uint4*)sW;
    uint4* sA4w = (uint4*)sA;
    int y0 = ty * TILE_R - 1, x0 = tx * TILE_C - 1;
#if HAVE_ASYNC_LDS
    for (int i = tid; i < 9216 / 8; i += 256)
        ASYNC_CP128(wsrc + i, sW4w + i);
    // stage 10x18 halo: async copy in-bounds, ds-store zeros on borders
    for (int i = tid; i < HALO_R * HALO_C * 4; i += 256) {
        int pix = i >> 2, q = i & 3;
        int rr = pix / HALO_C, cc = pix % HALO_C;
        int y = y0 + rr, xgl = x0 + cc;
        if (y >= 0 && y < HH && xgl >= 0 && xgl < WW) {
            uint4* g = (uint4*)(actIn + (((size_t)b * HH + y) * WW + xgl) * 32) + q;
            ASYNC_CP128(g, sA4w + pix * 4 + q);
        } else {
            sA4w[pix * 4 + q] = make_uint4(0u, 0u, 0u, 0u);
        }
    }
    if (tid < 64) sStats[tid] = 0.f;
    ASYNC_WAIT();
#else
    for (int i = tid; i < 9216 / 8; i += 256) sW4w[i] = wsrc[i];
    for (int i = tid; i < HALO_R * HALO_C * 4; i += 256) {
        int pix = i >> 2, q = i & 3;
        int rr = pix / HALO_C, cc = pix % HALO_C;
        int y = y0 + rr, xgl = x0 + cc;
        uint4 val = make_uint4(0u, 0u, 0u, 0u);
        if (y >= 0 && y < HH && xgl >= 0 && xgl < WW)
            val = ((const uint4*)(actIn + (((size_t)b * HH + y) * WW + xgl) * 32))[q];
        sA4w[pix * 4 + q] = val;
    }
    if (tid < 64) sStats[tid] = 0.f;
#endif
    __syncthreads();

    int wv = tid >> 5;           // wave -> output row in tile
    int lane = tid & 31;
    int m = lane & 15;           // M index (pixel column in tile)
    int hiHalf = lane >> 4;      // A: k0 = 0 or 8 ; C/D: M base 0 or 8

    v8f acc0 = {}; v8f acc1 = {};
    const uint4* sA4 = (const uint4*)sA;
    const uint4* sW4 = (const uint4*)sW;

#pragma unroll
    for (int t = 0; t < 9; ++t) {
        int dy = t / 3, dx = t % 3;
        // A fragment: lane L<16 -> K {0..7,16..23}; L>=16 -> K {8..15,24..31}
        int pix = (wv + dy) * HALO_C + (m + dx);
        int baseHalf = pix * 32 + hiHalf * 8;
        union { v16h v; uint4 q[2]; } A;
        A.q[0] = sA4[baseHalf >> 3];
        A.q[1] = sA4[(baseHalf + 16) >> 3];
        // B fragments (pre-swizzled): 32 contiguous bytes per lane
        union { v16h v; uint4 q[2]; } B0, B1;
        int b0off = ((t * 2 + 0) * 32 + lane) * 16;
        int b1off = ((t * 2 + 1) * 32 + lane) * 16;
        B0.q[0] = sW4[b0off >> 3]; B0.q[1] = sW4[(b0off >> 3) + 1];
        B1.q[0] = sW4[b1off >> 3]; B1.q[1] = sW4[(b1off >> 3) + 1];
        acc0 = __builtin_amdgcn_wmma_f32_16x16x32_f16(false, A.v, false, B0.v,
                                                      (short)0, acc0, false, false);
        acc1 = __builtin_amdgcn_wmma_f32_16x16x32_f16(false, A.v, false, B1.v,
                                                      (short)0, acc1, false, false);
    }

    // scatter C/D (lane L<16: N=L, M=r ; L>=16: N=L-16, M=r+8) + IN stats
    int nBase = lane & 15;
    int mBase = hiHalf ? 8 : 0;
    float s0 = 0.f, q0 = 0.f, s1 = 0.f, q1 = 0.f;
#pragma unroll
    for (int r = 0; r < 8; ++r) {
        float v0 = acc0[r], v1 = acc1[r];
        sOut[wv * 512 + (mBase + r) * 32 + nBase]      = v0;
        sOut[wv * 512 + (mBase + r) * 32 + 16 + nBase] = v1;
        s0 += v0; q0 += v0 * v0; s1 += v1; q1 += v1 * v1;
    }
    atomicAdd(&sStats[nBase], s0);
    atomicAdd(&sStats[32 + nBase], q0);
    atomicAdd(&sStats[16 + nBase], s1);
    atomicAdd(&sStats[48 + nBase], q1);
    __syncthreads();

    // coalesced f16 store of the 8x16x32 tile (each thread: 16 values = 32B)
    {
        int p = tid >> 1, halfsel = tid & 1;
        const float* src = &sOut[p * 32 + halfsel * 16];
        union { _Float16 h[16]; uint4 q[2]; } u;
#pragma unroll
        for (int j = 0; j < 16; ++j) u.h[j] = (_Float16)src[j];
        int pr = p >> 4, pc = p & 15;
        size_t o = ((((size_t)b * HH + (ty * TILE_R + pr)) * WW) + (tx * TILE_C + pc)) * 32
                   + halfsel * 16;
        uint4* dst = (uint4*)(actOut + o);
        dst[0] = u.q[0]; dst[1] = u.q[1];
    }
    if (tid < 64) atomicAdd(&stats[b * 64 + tid], sStats[tid]);
}

// ---------------------------------------------------------------------------
// InstanceNorm + ReLU, in place on channel-last f16 activations.
// ---------------------------------------------------------------------------
__global__ __launch_bounds__(256) void norm_kernel(_Float16* __restrict__ act,
                                                   const float* __restrict__ stats,
                                                   const float* __restrict__ gammas,
                                                   const float* __restrict__ betas,
                                                   int layer) {
    __shared__ float sMu[32], sRs[32], sG[32], sB[32];
    int b = blockIdx.y, tid = threadIdx.x;
    if (tid < 32) {
        float sum = stats[b * 64 + tid], sq = stats[b * 64 + 32 + tid];
        const float inv = 1.0f / (float)HW;
        float mu = sum * inv;
        float var = sq * inv - mu * mu;
        sMu[tid] = mu;
        sRs[tid] = rsqrtf(var + EPSV);
        sG[tid] = gammas[layer * 32 + tid];
        sB[tid] = betas[layer * 32 + tid];
    }
    __syncthreads();
    size_t base = ((size_t)b * HW + blockIdx.x * 256 + tid) * 32;
    uint4* pq = (uint4*)(act + base);
    union { _Float16 h[32]; uint4 q[4]; } u;
    u.q[0] = pq[0]; u.q[1] = pq[1]; u.q[2] = pq[2]; u.q[3] = pq[3];
#pragma unroll
    for (int c = 0; c < 32; ++c) {
        float v = (float)u.h[c];
        v = (v - sMu[c]) * sRs[c] * sG[c] + sB[c];
        u.h[c] = (_Float16)fmaxf(v, 0.f);
    }
    pq[0] = u.q[0]; pq[1] = u.q[1]; pq[2] = u.q[2]; pq[3] = u.q[3];
}

// ---------------------------------------------------------------------------
// Layer 5: 32->1 direct conv + residual -> pred (f32)
// ---------------------------------------------------------------------------
__global__ __launch_bounds__(256) void conv5_kernel(const _Float16* __restrict__ act,
                                                    const float* __restrict__ x,
                                                    const float* __restrict__ w5f,
                                                    float* __restrict__ pred) {
    __shared__ float sW[288];
    int b = blockIdx.y, tid = threadIdx.x;
    for (int i = tid; i < 288; i += 256) sW[i] = w5f[b * 288 + i];
    __syncthreads();
    int p = blockIdx.x * 256 + tid;
    int yy = p / WW, xx = p % WW;
    float a = 0.f;
#pragma unroll
    for (int t = 0; t < 9; ++t) {
        int dy = t / 3, dx = t % 3;
        int y2 = yy + dy - 1, x2 = xx + dx - 1;
        if (y2 >= 0 && y2 < HH && x2 >= 0 && x2 < WW) {
            const _Float16* row = act + (((size_t)b * HH + y2) * WW + x2) * 32;
#pragma unroll
            for (int ic = 0; ic < 32; ++ic) a += (float)row[ic] * sW[ic * 9 + t];
        }
    }
    pred[(size_t)b * HW + p] = a + x[(size_t)b * HW + p];
}

// ---------------------------------------------------------------------------
// DFT: twiddle table T[c*320+k] = exp(-2*pi*i*c*k/320), f32, L2-resident.
// Each pass: 1D transform of 8 rows per workgroup, transposed coalesced write.
// ---------------------------------------------------------------------------
__global__ void twiddle_kernel(float* __restrict__ twid) {
    int i = blockIdx.x * 256 + threadIdx.x;
    if (i >= HW) return;
    int c = i / WW, kk = i % WW;
    int ph = (c * kk) % 320;                       // exact integer phase
    float ang = -6.283185307179586f * (float)ph * (1.0f / 320.0f);
    float s, co;
    __sincosf(ang, &s, &co);
    ((float2*)twid)[i] = make_float2(co, s);
}

#define DFT_ROWS 8

template <bool REAL_IN, bool CONJ, bool REAL_OUT>
__global__ __launch_bounds__(320) void dft_kernel(const float* __restrict__ in,
                                                  float* __restrict__ out,
                                                  const float* __restrict__ twid) {
    __shared__ float2 rows[DFT_ROWS][320];         // 20.5 KB
    int r0 = blockIdx.x * DFT_ROWS;
    int b = blockIdx.y;
    int k = threadIdx.x;
#pragma unroll
    for (int j = 0; j < DFT_ROWS; ++j) {
        if (REAL_IN)
            rows[j][k] = make_float2(in[((size_t)b * 320 + r0 + j) * 320 + k], 0.f);
        else
            rows[j][k] = ((const float2*)in)[((size_t)b * 320 + r0 + j) * 320 + k];
    }
    __syncthreads();
    float ar[DFT_ROWS] = {}, ai[DFT_ROWS] = {};
    for (int c = 0; c < 320; ++c) {
        float2 w = ((const float2*)twid)[c * 320 + k];   // coalesced, 8x reused
        float wy = CONJ ? -w.y : w.y;
#pragma unroll
        for (int j = 0; j < DFT_ROWS; ++j) {
            float2 xv = rows[j][c];                      // LDS broadcast
            ar[j] += xv.x * w.x - xv.y * wy;
            ai[j] += xv.x * wy + xv.y * w.x;
        }
    }
    const float s = 0.05590169943749474f;                // 1/sqrt(320), ortho
    size_t o = ((size_t)b * 320 + k) * 320 + r0;         // transposed, contiguous in j
    if (REAL_OUT) {
#pragma unroll
        for (int j = 0; j < DFT_ROWS; ++j) out[o + j] = ar[j] * s;
    } else {
#pragma unroll
        for (int j = 0; j < DFT_ROWS; ++j)
            ((float2*)out)[o + j] = make_float2(ar[j] * s, ai[j] * s);
    }
}

// upd = mask*ks + (1-mask)*fft2(pred)   (all [b][u][v] layout)
__global__ void combine_kernel(const float* __restrict__ kp, const float* __restrict__ mask,
                               const float* __restrict__ kspace, float* __restrict__ outc) {
    int i = blockIdx.x * 256 + threadIdx.x;   // < B*HW
    float m = mask[i];
    float2 p = ((const float2*)kp)[i];
    float kr = kspace[2 * i], ki = kspace[2 * i + 1];
    ((float2*)outc)[i] = make_float2(m * kr + (1.f - m) * p.x,
                                     m * ki + (1.f - m) * p.y);
}

// ---------------------------------------------------------------------------
extern "C" void kernel_launch(void* const* d_in, const int* in_sizes, int n_in,
                              void* d_out, int out_size, void* d_ws, size_t ws_size,
                              hipStream_t stream) {
    const float* x      = (const float*)d_in[0];
    const float* kin    = (const float*)d_in[1];
    const float* acc    = (const float*)d_in[2];
    const float* mask   = (const float*)d_in[3];
    const float* W1     = (const float*)d_in[4];
    const float* b1     = (const float*)d_in[5];
    const float* Wmid   = (const float*)d_in[6];
    const float* bmid   = (const float*)d_in[7];
    const float* W5     = (const float*)d_in[8];
    const float* b5     = (const float*)d_in[9];
    const float* gammas = (const float*)d_in[10];
    const float* betas  = (const float*)d_in[11];
    float* out = (float*)d_out;

    char* wp = (char*)d_ws;
    auto alloc = [&](size_t bytes) -> char* {
        char* r = wp; wp += (bytes + 255) & ~(size_t)255; return r;
    };
    _Float16* actA  = (_Float16*)alloc((size_t)BSZ * HW * 32 * 2);   // 104.9 MB
    _Float16* actB  = (_Float16*)alloc((size_t)BSZ * HW * 32 * 2);   // 104.9 MB
    float*    stats = (float*)alloc(BSZ * 64 * 4);
    float*    w1f   = (float*)alloc(BSZ * 288 * 4);
    float*    w5f   = (float*)alloc(BSZ * 288 * 4);
    _Float16* wfrag = (_Float16*)alloc((size_t)3 * BSZ * 9216 * 2);  // 884 KB
    float*    pred  = (float*)alloc((size_t)BSZ * HW * 4);           // 6.6 MB
    float*    twid  = (float*)alloc((size_t)HW * 8);                 // 819 KB
    float*    cb1   = (float*)alloc((size_t)BSZ * HW * 8);           // 13.1 MB
    float*    cb2   = (float*)alloc((size_t)BSZ * HW * 8);           // 13.1 MB

    // weights + twiddles
    gen_small_kernel<<<(2 * BSZ * 288 + 255) / 256, 256, 0, stream>>>(acc, W1, b1, W5, b5, w1f, w5f);
    gen_frag_kernel<<<(3 * BSZ * 9216 + 255) / 256, 256, 0, stream>>>(acc, Wmid, bmid, wfrag);
    twiddle_kernel<<<(HW + 255) / 256, 256, 0, stream>>>(twid);

    dim3 pixGrid(HW / 256, BSZ);
    dim3 convGrid(WW / TILE_C, HH / TILE_R, BSZ);

    // layer 1
    (void)hipMemsetAsync(stats, 0, BSZ * 64 * 4, stream);
    conv1_kernel<<<pixGrid, 256, 0, stream>>>(x, actA, w1f, stats);
    norm_kernel<<<pixGrid, 256, 0, stream>>>(actA, stats, gammas, betas, 0);
    // mid layer 1: A -> B
    (void)hipMemsetAsync(stats, 0, BSZ * 64 * 4, stream);
    convmid_kernel<<<convGrid, 256, 0, stream>>>(actA, actB, wfrag + (size_t)0 * BSZ * 9216, stats);
    norm_kernel<<<pixGrid, 256, 0, stream>>>(actB, stats, gammas, betas, 1);
    // mid layer 2: B -> A
    (void)hipMemsetAsync(stats, 0, BSZ * 64 * 4, stream);
    convmid_kernel<<<convGrid, 256, 0, stream>>>(actB, actA, wfrag + (size_t)1 * BSZ * 9216, stats);
    norm_kernel<<<pixGrid, 256, 0, stream>>>(actA, stats, gammas, betas, 2);
    // mid layer 3: A -> B
    (void)hipMemsetAsync(stats, 0, BSZ * 64 * 4, stream);
    convmid_kernel<<<convGrid, 256, 0, stream>>>(actA, actB, wfrag + (size_t)2 * BSZ * 9216, stats);
    norm_kernel<<<pixGrid, 256, 0, stream>>>(actB, stats, gammas, betas, 3);
    // layer 5 + residual
    conv5_kernel<<<pixGrid, 256, 0, stream>>>(actB, x, w5f, pred);

    // fft2 (ortho) -> combine -> ifft2 (ortho) -> real
    dim3 dftGrid(320 / DFT_ROWS, BSZ);
    dft_kernel<true,  false, false><<<dftGrid, 320, 0, stream>>>(pred, cb1, twid); // rows of pred
    dft_kernel<false, false, false><<<dftGrid, 320, 0, stream>>>(cb1, cb2, twid);  // -> KP[b][u][v]
    combine_kernel<<<(BSZ * HW) / 256, 256, 0, stream>>>(cb2, mask, kin, cb1);     // -> U[b][u][v]
    dft_kernel<false, true,  false><<<dftGrid, 320, 0, stream>>>(cb1, cb2, twid);  // inverse pass 1
    dft_kernel<false, true,  true ><<<dftGrid, 320, 0, stream>>>(cb2, out, twid);  // inverse pass 2 -> real
}
